// DynaMix_8546984919177
// MI455X (gfx1250) — matrix operand order repaired and checked by reading, not compiled, not run on hardware
//
#include <hip/hip_runtime.h>
#include <hip/hip_bf16.h>

// ---------------------------------------------------------------------------
// DynaMix step for MI455X (gfx1250): single NT streaming pass over context
// with online softmax, tiny gating MLP, WMMA bf16 expert mixture.
// ---------------------------------------------------------------------------

#define Mdim 64
#define Ndim 32
#define Edim 16
#define Bdim 1024
#define Sdim 2048
#define LOG2E 1.4426950408889634f

typedef float f4 __attribute__((ext_vector_type(4)));
typedef __attribute__((ext_vector_type(16))) __bf16 v16bf;
typedef __attribute__((ext_vector_type(8)))  float  v8f;

// ---------------------------------------------------------------------------
// Kernel 1: z_obs = D @ z   (32 x 1024). Tiny.
// ---------------------------------------------------------------------------
__global__ __launch_bounds__(256) void zobs_kernel(const float* __restrict__ D,
                                                   const float* __restrict__ z,
                                                   float* __restrict__ zobs) {
  int t = blockIdx.x * 256 + threadIdx.x;        // 0..32767
  int i = t >> 10;
  int b = t & 1023;
  float acc = 0.f;
#pragma unroll
  for (int m = 0; m < Mdim; ++m) acc += D[i * Mdim + m] * z[m * Bdim + b];
  zobs[i * Bdim + b] = acc;
}

// ---------------------------------------------------------------------------
// Kernel 2: streaming online-softmax attention over context.
// One wave handles 4 consecutive b's (lane = b_sub*8 + feat_quad) over one
// 256-step S-chunk. 512B coalesced b128 NT loads + prefetch 4 rows ahead.
// Outputs per (b,chunk): running max m, sum l, weighted sums c0[32], c1[32].
// ---------------------------------------------------------------------------
__global__ __launch_bounds__(256) void attn_stream(const float* __restrict__ ctx,
                                                   const float* __restrict__ zobs,
                                                   const float* __restrict__ temp1,
                                                   float* __restrict__ pm,
                                                   float* __restrict__ pl,
                                                   float* __restrict__ pc0,
                                                   float* __restrict__ pc1) {
  const int warp = threadIdx.x >> 5;
  const int lane = threadIdx.x & 31;
  const int g = blockIdx.x * 8 + warp;   // 0..2047
  const int bg = g >> 3;                 // b-group 0..255
  const int chunk = g & 7;               // S-chunk 0..7
  const int b0 = bg * 4;
  const int myb = b0 + (lane >> 3);
  const int f0 = (lane & 7) * 4;
  const int s0 = chunk * 256;
  const int s1 = (s0 + 256 < Sdim - 1) ? (s0 + 256) : (Sdim - 1);  // steps in [s0,s1)

  const float t1 = fmaxf(fabsf(temp1[0]), 1e-4f);
  const float nscale = LOG2E / t1;

  f4 zo;
  zo.x = zobs[(f0 + 0) * Bdim + myb];
  zo.y = zobs[(f0 + 1) * Bdim + myb];
  zo.z = zobs[(f0 + 2) * Bdim + myb];
  zo.w = zobs[(f0 + 3) * Bdim + myb];

  const size_t rowF4 = (size_t)Bdim * Ndim / 4;  // f4 elements per s-step
  const f4* p = (const f4*)(ctx) + (size_t)s0 * rowF4 + (size_t)(b0 * Ndim) / 4 + lane;

  f4 cur = __builtin_nontemporal_load(p);
  p += rowF4;  // now points at row s0+1

  float m = -INFINITY, l = 0.f;
  f4 a0 = {0.f, 0.f, 0.f, 0.f};
  f4 a1 = {0.f, 0.f, 0.f, 0.f};

  for (int s = s0; s < s1; ++s) {
    if (s + 5 <= Sdim - 1) __builtin_prefetch(p + 4 * rowF4, 0, 0);   // global_prefetch_b8
    f4 nxt = __builtin_nontemporal_load(p);
    p += rowF4;

    float r = fabsf(cur.x - zo.x) + fabsf(cur.y - zo.y) +
              fabsf(cur.z - zo.z) + fabsf(cur.w - zo.w);
    r += __shfl_xor(r, 1);
    r += __shfl_xor(r, 2);
    r += __shfl_xor(r, 4);                 // L1 distance for this b (uniform in 8-lane group)

    float lg = -r * nscale;                // logit in log2 domain
    float mn = fmaxf(m, lg);
    float corr = exp2f(m - mn);
    float pw = exp2f(lg - mn);
    l = l * corr + pw;
    a0 = a0 * corr + pw * cur;
    a1 = a1 * corr + pw * nxt;
    m = mn;
    cur = nxt;
  }

  const int idx = myb * 8 + chunk;
  if ((lane & 7) == 0) { pm[idx] = m; pl[idx] = l; }
  ((f4*)pc0)[idx * 8 + (lane & 7)] = a0;
  ((f4*)pc1)[idx * 8 + (lane & 7)] = a1;
}

// ---------------------------------------------------------------------------
// Kernel 3: merge chunk partials (log-sum-exp), embedding (32x32 conv fold),
// gating MLP (96->16 relu ->16), softmax over E. One wave per b.
// ---------------------------------------------------------------------------
__global__ __launch_bounds__(256) void gate_kernel(const float* __restrict__ z,
                                                   const float* __restrict__ conv_w,
                                                   const float* __restrict__ conv_b,
                                                   const float* __restrict__ W1,
                                                   const float* __restrict__ b1,
                                                   const float* __restrict__ W2,
                                                   const float* __restrict__ b2,
                                                   const float* __restrict__ temp2,
                                                   const float* __restrict__ pm,
                                                   const float* __restrict__ pl,
                                                   const float* __restrict__ pc0,
                                                   const float* __restrict__ pc1,
                                                   float* __restrict__ wexp) {
  __shared__ float sC0[8][32];
  __shared__ float sC1[8][32];
  __shared__ float sComb[8][96];
  __shared__ float sH[8][16];

  const int wv = threadIdx.x >> 5;
  const int lane = threadIdx.x & 31;
  const int b = blockIdx.x * 8 + wv;     // 0..1023

  // merge 8 chunk partials
  float mc[8];
  float Mg = -INFINITY;
#pragma unroll
  for (int c = 0; c < 8; ++c) { mc[c] = pm[b * 8 + c]; Mg = fmaxf(Mg, mc[c]); }
  float L = 0.f, c0t = 0.f, c1t = 0.f;
#pragma unroll
  for (int c = 0; c < 8; ++c) {
    float cf = exp2f(mc[c] - Mg);
    L += cf * pl[b * 8 + c];
    c0t += cf * pc0[(b * 8 + c) * 32 + lane];
    c1t += cf * pc1[(b * 8 + c) * 32 + lane];
  }
  float inv = 1.f / L;
  sC0[wv][lane] = c0t * inv;
  sC1[wv][lane] = c1t * inv;
  __syncthreads();

  // embedding[o] = sum_i c0n[i]*w[o,i,0] + c1n[i]*w[o,i,1] + cb[o]; lane = o
  float emb = conv_b[lane];
#pragma unroll
  for (int i = 0; i < 32; ++i)
    emb += sC0[wv][i] * conv_w[lane * 64 + i * 2] + sC1[wv][i] * conv_w[lane * 64 + i * 2 + 1];
  sComb[wv][lane] = emb;
  sComb[wv][32 + lane] = z[lane * Bdim + b];
  sComb[wv][64 + lane] = z[(32 + lane) * Bdim + b];
  __syncthreads();

  if (lane < 16) {
    float hsum = b1[lane];
#pragma unroll
    for (int j = 0; j < 96; ++j) hsum += sComb[wv][j] * W1[lane * 96 + j];
    sH[wv][lane] = fmaxf(hsum, 0.f);
  }
  __syncthreads();

  if (lane < 16) {
    float msum = b2[lane];
#pragma unroll
    for (int k = 0; k < 16; ++k) msum += sH[wv][k] * W2[lane * 16 + k];
    const float t2 = fmaxf(fabsf(temp2[0]), 1e-4f);
    float v = -msum * (LOG2E / t2);
    float vmax = v;
    vmax = fmaxf(vmax, __shfl_xor(vmax, 1));
    vmax = fmaxf(vmax, __shfl_xor(vmax, 2));
    vmax = fmaxf(vmax, __shfl_xor(vmax, 4));
    vmax = fmaxf(vmax, __shfl_xor(vmax, 8));
    float pp = exp2f(v - vmax);
    float ps = pp;
    ps += __shfl_xor(ps, 1);
    ps += __shfl_xor(ps, 2);
    ps += __shfl_xor(ps, 4);
    ps += __shfl_xor(ps, 8);
    wexp[lane * Bdim + b] = pp / ps;
  }
}

// ---------------------------------------------------------------------------
// Kernel 4: expert mixture via v_wmma_f32_16x16x32_bf16.
// out[m,b] = sum_e w[e,b]*Wz[e,m,b] + z[m,b]*(sum_e w A[e,m]) + sum_e w h[e,m]
// One wave per 16-column b-tile; 4 m-tiles x 2 k-steps x 16 experts.
// ---------------------------------------------------------------------------
__device__ inline v16bf load_a_tile(const float* __restrict__ We, int mt, int kt,
                                    int ln, int laneHi) {
  const int m = mt * 16 + ln;
  const int kbase = kt * 32 + laneHi * 8;
  v16bf a;
#pragma unroll
  for (int j = 0; j < 16; ++j) {
    int k = kbase + (j < 8 ? j : j + 8);
    a[j] = (__bf16)We[m * Mdim + k];
  }
  return a;
}

__global__ __launch_bounds__(128) void expert_mix(const float* __restrict__ z,
                                                  const float* __restrict__ A,
                                                  const float* __restrict__ W,
                                                  const float* __restrict__ h,
                                                  const float* __restrict__ wexp,
                                                  float* __restrict__ out) {
  const int lane = threadIdx.x & 31;
  const int wv = threadIdx.x >> 5;
  const int bt = blockIdx.x * 4 + wv;    // 0..63
  const int laneHi = lane >> 4;
  const int ln = lane & 15;
  const int bcol = bt * 16 + ln;         // this lane's batch column (B/C layout)

  // B operand: zcat tile (64 x 16) -> two 32x16 bf16 k-slices
  v16bf Bt[2];
#pragma unroll
  for (int kt = 0; kt < 2; ++kt) {
    const int kbase = kt * 32 + laneHi * 8;
#pragma unroll
    for (int j = 0; j < 16; ++j) {
      int k = kbase + (j < 8 ? j : j + 8);
      float v = z[k * Bdim + bcol];
      if (k >= Mdim - 2) v = fmaxf(v, 0.f);   // zcat = [z[:62]; relu(z[62:])]
      Bt[kt][j] = (__bf16)v;
    }
  }

  // per-column expert weights
  float wcol[Edim];
#pragma unroll
  for (int e = 0; e < Edim; ++e) wcol[e] = wexp[e * Bdim + bcol];

#pragma unroll
  for (int mt = 0; mt < 4; ++mt) {
    v8f outAcc = {0.f, 0.f, 0.f, 0.f, 0.f, 0.f, 0.f, 0.f};
#pragma unroll
    for (int e = 0; e < Edim; ++e) {
      const float* We = W + e * (Mdim * Mdim);
      v16bf aA = load_a_tile(We, mt, 0, ln, laneHi);
      v16bf aB = load_a_tile(We, mt, 1, ln, laneHi);
      v8f cz = {0.f, 0.f, 0.f, 0.f, 0.f, 0.f, 0.f, 0.f};
      v8f c = __builtin_amdgcn_wmma_f32_16x16x32_bf16(false, aA, false, Bt[0],
                                                      (short)0, cz, false, false);
      c = __builtin_amdgcn_wmma_f32_16x16x32_bf16(false, aB, false, Bt[1],
                                                  (short)0, c, false, false);
      const float we = wcol[e];
#pragma unroll
      for (int r = 0; r < 8; ++r) outAcc[r] += we * c[r];
    }
    // epilogue: + z*Aw + Hw, write out
#pragma unroll
    for (int r = 0; r < 8; ++r) {
      const int mrow = mt * 16 + laneHi * 8 + r;
      float aw = 0.f, hw = 0.f;
#pragma unroll
      for (int e = 0; e < Edim; ++e) {
        float we = wcol[e];
        aw += we * A[e * Mdim + mrow];
        hw += we * h[e * Mdim + mrow];
      }
      float zv = z[mrow * Bdim + bcol];
      out[mrow * Bdim + bcol] = outAcc[r] + zv * aw + hw;
    }
  }
}

// ---------------------------------------------------------------------------
extern "C" void kernel_launch(void* const* d_in, const int* in_sizes, int n_in,
                              void* d_out, int out_size, void* d_ws, size_t ws_size,
                              hipStream_t stream) {
  (void)in_sizes; (void)n_in; (void)out_size; (void)ws_size;

  const float* z      = (const float*)d_in[0];
  const float* ctx    = (const float*)d_in[1];
  const float* conv_w = (const float*)d_in[2];
  const float* conv_b = (const float*)d_in[3];
  const float* temp1  = (const float*)d_in[4];
  const float* Dm     = (const float*)d_in[5];
  const float* W1     = (const float*)d_in[6];
  const float* b1     = (const float*)d_in[7];
  const float* W2     = (const float*)d_in[8];
  const float* b2     = (const float*)d_in[9];
  const float* temp2  = (const float*)d_in[10];
  const float* A      = (const float*)d_in[11];
  const float* W      = (const float*)d_in[12];
  const float* h      = (const float*)d_in[13];

  float* ws   = (float*)d_ws;
  float* zobs = ws;                    // 32*1024            = 32768
  float* pm   = zobs + 32768;          // 1024*8             = 8192
  float* pl   = pm + 8192;             // 1024*8             = 8192
  float* pc0  = pl + 8192;             // 1024*8*32          = 262144
  float* pc1  = pc0 + 262144;          // 1024*8*32          = 262144
  float* wexp = pc1 + 262144;          // 16*1024            = 16384

  zobs_kernel<<<128, 256, 0, stream>>>(Dm, z, zobs);
  attn_stream<<<256, 256, 0, stream>>>(ctx, zobs, temp1, pm, pl, pc0, pc1);
  gate_kernel<<<128, 256, 0, stream>>>(z, conv_w, conv_b, W1, b1, W2, b2, temp2,
                                       pm, pl, pc0, pc1, wexp);
  expert_mix<<<16, 128, 0, stream>>>(z, A, W, h, wexp, (float*)d_out);
}
